// LSTM_73770358276747
// MI455X (gfx1250) — compile-verified, matrix operand order checked by reading, not run on hardware
//
#include <hip/hip_runtime.h>
#include <hip/hip_bf16.h>
#include <stdint.h>

#define DEV __device__ __forceinline__

typedef __bf16 v16bf __attribute__((ext_vector_type(16)));
typedef float  v8f   __attribute__((ext_vector_type(8)));

static constexpr int Bb = 32, Tt = 512, Hh = 1024, Gg = 4096;

// packed sizes in 32-bit words
static constexpr size_t W4_WORDS = 256ull * 32 * 32 * 8;  // 2M words (8MB)
static constexpr size_t WG_WORDS = 64ull * 64 * 32 * 8;   // 1M words (4MB)
static constexpr size_t XP_WORDS = 1024ull * 32 * 32 * 8; // 8M words (32MB)

// workspace layout (bytes)
static constexpr size_t SZ_XG    = (size_t)Tt * Bb * Gg * 4;   // 256MB (fragment-packed)
static constexpr size_t SZ_H32   = (size_t)Tt * Bb * Hh * 4;   // 64MB
static constexpr size_t SZ_HBF   = (size_t)Tt * Bb * Hh * 2;   // 32MB
static constexpr size_t OFF_XG_F  = 0;
static constexpr size_t OFF_XG_B  = OFF_XG_F + SZ_XG;
static constexpr size_t OFF_HF32  = OFF_XG_B + SZ_XG;
static constexpr size_t OFF_HB32  = OFF_HF32 + SZ_H32;
static constexpr size_t OFF_HFBF  = OFF_HB32 + SZ_H32;
static constexpr size_t OFF_HBBF  = OFF_HFBF + SZ_HBF;
static constexpr size_t OFF_XBF   = OFF_HBBF + SZ_HBF;         // x packed, 32MB
static constexpr size_t OFF_WIH_F = OFF_XBF + XP_WORDS * 4;
static constexpr size_t OFF_WIH_B = OFF_WIH_F + W4_WORDS * 4;
static constexpr size_t OFF_WHH_F = OFF_WIH_B + W4_WORDS * 4;
static constexpr size_t OFF_WHH_B = OFF_WHH_F + W4_WORDS * 4;
static constexpr size_t OFF_WG_P  = OFF_WHH_B + W4_WORDS * 4;
static constexpr size_t OFF_BIAS  = OFF_WG_P + WG_WORDS * 4;       // 2*4096 fp32
static constexpr size_t OFF_HCUR  = OFF_BIAS + 2 * Gg * 4;         // 2*32*1024 bf16
static constexpr size_t OFF_CCUR  = OFF_HCUR + 2ull * Bb * Hh * 2; // 2*32*1024 fp32
static constexpr size_t OFF_BARS  = OFF_CCUR + 2ull * Bb * Hh * 4; // 32 u32

// ---- helpers -------------------------------------------------------------

// K index of the bf16 pair held in VGPR word v for lane-half khalf (16-bit A/B
// fragment layout, CDNA5 ISA 7.12.2)
DEV int kbase(int khalf, int v) {
    return ((v & 4) ? 16 : 0) + ((v & 3) << 1) + (khalf << 3);
}

DEV unsigned short f2bf(float f) {
    unsigned u = __float_as_uint(f);
    unsigned r = u + 0x7FFFu + ((u >> 16) & 1u);
    return (unsigned short)(r >> 16);
}
DEV unsigned packbf(float a, float b) {
    return (unsigned)f2bf(a) | ((unsigned)f2bf(b) << 16);
}

DEV v8f wmma_bf16(uint4 alo, uint4 ahi, uint4 blo, uint4 bhi, v8f c) {
    union { v16bf v; uint4 q[2]; } A, Bm;
    A.q[0] = alo; A.q[1] = ahi; Bm.q[0] = blo; Bm.q[1] = bhi;
    return __builtin_amdgcn_wmma_f32_16x16x32_bf16(
        /*neg_a=*/false, A.v, /*neg_b=*/false, Bm.v,
        /*c_mod=*/(short)0, c, /*reuse_a=*/false, /*reuse_b=*/false);
}

DEV float sigmoidf(float x) { return 1.0f / (1.0f + __expf(-x)); }

// low 32 bits of a generic pointer to LDS == DS byte address
DEV unsigned lds_addr32(const void* p) { return (unsigned)(uintptr_t)p; }

// async direct memory->LDS copy of 16 bytes (ASYNCcnt-tracked)
DEV void async_ld_b128(unsigned lds_off, const void* gsrc) {
    asm volatile("global_load_async_to_lds_b128 %0, %1, off"
                 :: "v"(lds_off), "v"((unsigned long long)(uintptr_t)gsrc)
                 : "memory");
}
DEV void wait_async0() {
    asm volatile("s_wait_asynccnt 0x0" ::: "memory");
}

// ---- init ----------------------------------------------------------------

__global__ __launch_bounds__(256) void init_kernel(unsigned* hcur_w, unsigned* ccur_w,
                                                   unsigned* bars) {
    int i = blockIdx.x * 256 + threadIdx.x;
    if (i < 2 * Bb * Hh / 2) hcur_w[i] = 0u;
    if (i < 2 * Bb * Hh)     ccur_w[i] = 0u;
    if (i < 32)              bars[i] = 0u;
}

// ---- packing -------------------------------------------------------------
// W4 dst: [ntile(256)][kstep(32)][lane][8w]; gate-permuted row n' = j*4+gate.
__global__ __launch_bounds__(256) void pack_w4_kernel(const float* __restrict__ W,
                                                      unsigned* __restrict__ dst) {
    int widx = blockIdx.x * 256 + threadIdx.x;      // < 2097152
    int v = widx & 7, ln = (widx >> 3) & 31, ks = (widx >> 8) & 31, ntile = widx >> 13;
    int np = ntile * 16 + (ln & 15);
    int row = (np & 3) * Hh + (np >> 2);
    int k0 = ks * 32 + kbase(ln >> 4, v);
    const float* wr = W + (size_t)row * Hh + k0;
    dst[widx] = packbf(wr[0], wr[1]);
}

__global__ __launch_bounds__(256) void pack_wg_kernel(const float* __restrict__ W,
                                                      unsigned* __restrict__ dst) {
    int widx = blockIdx.x * 256 + threadIdx.x;      // < 1048576
    int v = widx & 7, ln = (widx >> 3) & 31, ks = (widx >> 8) & 63, ntile = widx >> 14;
    int n = ntile * 16 + (ln & 15);
    int k0 = ks * 32 + kbase(ln >> 4, v);
    const float* wr = W + (size_t)n * (2 * Hh) + k0;
    dst[widx] = packbf(wr[0], wr[1]);
}

// x -> bf16 A fragments: [rtile(1024)][kstep(32)][lane][8w]; rows r = t*32+b.
__global__ __launch_bounds__(256) void pack_x_kernel(const float* __restrict__ x,
                                                     unsigned* __restrict__ dst) {
    int widx = blockIdx.x * 256 + threadIdx.x;      // < 8388608
    int v = widx & 7, ln = (widx >> 3) & 31, ks = (widx >> 8) & 31, rtile = widx >> 13;
    int m = ln & 15, khalf = ln >> 4;
    int r = rtile * 16 + m, t = r >> 5, b = r & 31;
    int k0 = ks * 32 + kbase(khalf, v);
    const float* xr = x + ((size_t)b * Tt + t) * Hh + k0;
    dst[widx] = packbf(xr[0], xr[1]);
}

__global__ __launch_bounds__(256) void bias_kernel(const float* bih_f, const float* bhh_f,
                                                   const float* bih_b, const float* bhh_b,
                                                   float* biasp) {
    int i = blockIdx.x * 256 + threadIdx.x;         // < 8192
    int dir = i >> 12;
    int np = i & 4095;
    int row = (np & 3) * Hh + (np >> 2);
    const float* a = dir ? bih_b : bih_f;
    const float* b = dir ? bhh_b : bhh_f;
    biasp[i] = a[row] + b[row];
}

// ---- input GEMM: xg = x @ W_ih'^T + bias' --------------------------------
// grid (1024 rtiles, 8, 2 dirs), block 256; wave = 1 rtile x 4 ntiles.
// xg stored fragment-packed: [t][ntile(256)][mtile(2)][lane][8 floats].
__global__ __launch_bounds__(256) void input_gemm_kernel(
    const unsigned* __restrict__ xpack,
    const unsigned* __restrict__ wp_f, const unsigned* __restrict__ wp_b,
    const float* __restrict__ biasp,
    float* __restrict__ xg_f, float* __restrict__ xg_b)
{
    int wid = threadIdx.x >> 5, lane = threadIdx.x & 31;
    int rtile = blockIdx.x;
    int ngrp = blockIdx.y * 8 + wid;                // 0..63 -> ntiles [4g,4g+4)
    int dir = blockIdx.z;
    const unsigned* wp = dir ? wp_b : wp_f;
    float* xg = dir ? xg_b : xg_f;
    const float* bias = biasp + dir * Gg;
    int ncol = lane & 15;

    v8f acc[4];
#pragma unroll
    for (int j = 0; j < 4; ++j) {
        float bv = bias[(ngrp * 4 + j) * 16 + ncol];
#pragma unroll
        for (int q = 0; q < 8; ++q) acc[j][q] = bv;
    }

    const uint4* ap = (const uint4*)(xpack + (((size_t)rtile * 32) * 32 + lane) * 8);
    for (int ks = 0; ks < 32; ++ks) {
        uint4 alo = ap[0], ahi = ap[1];
        ap += 64;                                   // next kstep (32 lanes * 8w / 4)
        uint4 blo[4], bhi[4];
#pragma unroll
        for (int j = 0; j < 4; ++j) {
            const uint4* bp = (const uint4*)(wp +
                (((size_t)(ngrp * 4 + j) * 32 + ks) * 32 + lane) * 8);
            blo[j] = bp[0]; bhi[j] = bp[1];
        }
#pragma unroll
        for (int j = 0; j < 4; ++j)
            acc[j] = wmma_bf16(alo, ahi, blo[j], bhi[j], acc[j]);
    }

    int t2 = rtile >> 1, mtile = rtile & 1;
#pragma unroll
    for (int j = 0; j < 4; ++j) {
        int ntile = ngrp * 4 + j;
        union { v8f v; float4 f4[2]; } u; u.v = acc[j];
        float* dst = xg + ((((size_t)t2 * 256 + ntile) * 2 + mtile) * 32 + lane) * 8;
        *(float4*)dst       = u.f4[0];
        *(float4*)(dst + 4) = u.f4[1];
    }
}

// ---- recurrent scan ------------------------------------------------------
// grid (32 blocks, 2 dirs), block 256 (8 waves).  Wave w owns gate-tile
// ntile = blockIdx.x*8+w -> hidden columns [ntile*4, ntile*4+4) (all gates).
__global__ __launch_bounds__(256) void scan_kernel(
    const float* __restrict__ xg_f, const float* __restrict__ xg_b,
    const unsigned* __restrict__ whh_f, const unsigned* __restrict__ whh_b,
    unsigned short* __restrict__ hcur, float* __restrict__ ccur,
    float* __restrict__ hf32, float* __restrict__ hb32,
    unsigned short* __restrict__ hfbf, unsigned short* __restrict__ hbbf,
    unsigned* __restrict__ bars)
{
    __shared__ unsigned lds[16384];                 // 64KB: h fragments / gate staging
    int tid = threadIdx.x;
    int wid = tid >> 5, lane = tid & 31;
    int dir = blockIdx.y;
    const float* xg = dir ? xg_b : xg_f;
    const unsigned* wp = dir ? whh_b : whh_f;
    unsigned short* hc = hcur + (size_t)dir * Bb * Hh;
    float* cc = ccur + (size_t)dir * Bb * Hh;
    float* h32 = dir ? hb32 : hf32;
    unsigned short* hbf = dir ? hbbf : hfbf;
    unsigned* bar = bars + dir * 16;
    int wglob = blockIdx.x * 8 + wid;               // ntile 0..255
    int ncol = lane & 15, khalf = lane >> 4;
    unsigned ldsbase = lds_addr32(&lds[0]);

    for (int s = 0; s < Tt; ++s) {
        int t = dir ? (Tt - 1 - s) : s;

        // async-stage h (bf16, 32x1024) into LDS pre-packed as A fragments:
        // group g = word widx>>2; 16B contiguous on both sides.
        const unsigned* hw = (const unsigned*)hc;
#pragma unroll 4
        for (int i = 0; i < 16; ++i) {
            int g = tid + i * 256;                  // < 4096
            int vh = g & 1, ln = (g >> 1) & 31, ks2 = (g >> 6) & 31, mt = g >> 11;
            int mm = mt * 16 + (ln & 15), kh = ln >> 4;
            const unsigned* src = hw + mm * (Hh / 2) + ks2 * 16 + kh * 4 + vh * 8;
            async_ld_b128(ldsbase + (unsigned)g * 16, src);
        }
        wait_async0();
        __syncthreads();

        // C init = xg[t] fragment (input contribution + bias), packed layout
        const float* xgt = xg + (size_t)t * (Bb * Gg);
        union { v8f v; float4 f4[2]; } a0u, a1u;
        {
            const float* p0 = xgt + (((size_t)wglob * 2 + 0) * 32 + lane) * 8;
            const float* p1 = xgt + (((size_t)wglob * 2 + 1) * 32 + lane) * 8;
            a0u.f4[0] = *(const float4*)p0; a0u.f4[1] = *(const float4*)(p0 + 4);
            a1u.f4[0] = *(const float4*)p1; a1u.f4[1] = *(const float4*)(p1 + 4);
        }
        v8f acc0 = a0u.v, acc1 = a1u.v;

        for (int ks = 0; ks < 32; ++ks) {
            const uint4* a0p = (const uint4*)&lds[((size_t)(0 * 32 + ks) * 32 + lane) * 8];
            const uint4* a1p = (const uint4*)&lds[((size_t)(1 * 32 + ks) * 32 + lane) * 8];
            const uint4* bp  = (const uint4*)(wp + (((size_t)wglob * 32 + ks) * 32 + lane) * 8);
            uint4 blo = bp[0], bhi = bp[1];
            acc0 = wmma_bf16(a0p[0], a0p[1], blo, bhi, acc0);
            acc1 = wmma_bf16(a1p[0], a1p[1], blo, bhi, acc1);
        }
        __syncthreads();

        // stage gate pre-activations: stage[wid][mtile][mloc][nloc]
        float* gst = (float*)lds;
        int gbase = wid * 512;
#pragma unroll
        for (int q = 0; q < 8; ++q) {
            gst[gbase + 0 * 256 + (q + 8 * khalf) * 16 + ncol] = acc0[q];
            gst[gbase + 1 * 256 + (q + 8 * khalf) * 16 + ncol] = acc1[q];
        }
        __syncthreads();

        // wave-local LSTM elementwise: lane = batch m, 4 hidden columns
        int m = lane;
#pragma unroll
        for (int jl = 0; jl < 4; ++jl) {
            int idx = gbase + (m >> 4) * 256 + (m & 15) * 16 + jl * 4;
            float gi = gst[idx + 0], gf = gst[idx + 1];
            float gg = gst[idx + 2], go = gst[idx + 3];
            int j = wglob * 4 + jl;
            size_t ci = (size_t)m * Hh + j;
            float cold = cc[ci];
            float cn = sigmoidf(gf) * cold + sigmoidf(gi) * tanhf(gg);
            float hv = sigmoidf(go) * tanhf(cn);
            cc[ci] = cn;
            size_t hi = ((size_t)t * Bb + m) * Hh + j;
            h32[hi] = hv;
            unsigned short h16 = f2bf(hv);
            hbf[hi] = h16;
            hc[ci] = h16;
        }

        // one grid-wide barrier per step (per direction, 32 blocks)
        __threadfence();
        __syncthreads();
        if (tid == 0) {
            unsigned gen = ((volatile unsigned*)bar)[1];
            unsigned prev = atomicAdd(&bar[0], 1u);
            if (prev == 31u) {
                ((volatile unsigned*)bar)[0] = 0u;
                __threadfence();
                atomicAdd(&bar[1], 1u);
            } else {
                while (((volatile unsigned*)bar)[1] == gen) __builtin_amdgcn_s_sleep(1);
            }
        }
        __syncthreads();
        __threadfence();
    }
}

// ---- final gate: out = sigma*hf + (1-sigma)*hb ---------------------------
// grid (1024 rtiles, 8), block 256; A tile (16 rows x K=2048) async-staged to
// LDS once per block and shared by all 8 waves (8 ntiles).
__global__ __launch_bounds__(256) void gate_kernel(
    const unsigned short* __restrict__ hfbf, const unsigned short* __restrict__ hbbf,
    const unsigned* __restrict__ wgp, const float* __restrict__ bg,
    const float* __restrict__ hf32, const float* __restrict__ hb32,
    float* __restrict__ out)
{
    __shared__ unsigned alds[16384];                // 64KB: [ks(64)][lane][8w]
    int tid = threadIdx.x;
    int wid = tid >> 5, lane = tid & 31;
    int rtile = blockIdx.x;
    int ntile = blockIdx.y * 8 + wid;               // 0..63
    int ncol = lane & 15, khalf = lane >> 4;
    const unsigned* hfw = (const unsigned*)hfbf;
    const unsigned* hbw = (const unsigned*)hbbf;
    int n = ntile * 16 + ncol;
    unsigned ldsbase = lds_addr32(&alds[0]);

    // async-stage A fragments for this rtile (hf rows for ks<32, hb after)
#pragma unroll 4
    for (int i = 0; i < 16; ++i) {
        int g = tid + i * 256;                      // < 4096
        int vh = g & 1, ln = (g >> 1) & 31, ks = g >> 6;   // ks 0..63
        int m = ln & 15, kh = ln >> 4;
        int row = rtile * 16 + m;
        const unsigned* base = (ks < 32) ? hfw : hbw;
        const unsigned* src = base + (size_t)row * (Hh / 2) + (ks & 31) * 16 + kh * 4 + vh * 8;
        async_ld_b128(ldsbase + (unsigned)g * 16, src);
    }
    wait_async0();
    __syncthreads();

    float bv = bg[n];
    v8f acc;
#pragma unroll
    for (int q = 0; q < 8; ++q) acc[q] = bv;

    for (int ks = 0; ks < 64; ++ks) {
        const uint4* apv = (const uint4*)&alds[((size_t)ks * 32 + lane) * 8];
        const uint4* bp = (const uint4*)(wgp + (((size_t)ntile * 64 + ks) * 32 + lane) * 8);
        acc = wmma_bf16(apv[0], apv[1], bp[0], bp[1], acc);
    }
#pragma unroll
    for (int q = 0; q < 8; ++q) {
        int row = rtile * 16 + q + 8 * khalf;
        int t = row >> 5, b = row & 31;
        float sgm = sigmoidf(acc[q]);
        float hf = hf32[(size_t)row * Hh + n];
        float hb = hb32[(size_t)row * Hh + n];
        out[((size_t)b * Tt + t) * Hh + n] = sgm * hf + (1.0f - sgm) * hb;
    }
}

// ---- launch --------------------------------------------------------------

extern "C" void kernel_launch(void* const* d_in, const int* in_sizes, int n_in,
                              void* d_out, int out_size, void* d_ws, size_t ws_size,
                              hipStream_t stream) {
    (void)in_sizes; (void)n_in; (void)out_size; (void)ws_size;

    const float* x     = (const float*)d_in[0];
    const float* Wih_f = (const float*)d_in[1];
    const float* Whh_f = (const float*)d_in[2];
    const float* bih_f = (const float*)d_in[3];
    const float* bhh_f = (const float*)d_in[4];
    const float* Wih_b = (const float*)d_in[5];
    const float* Whh_b = (const float*)d_in[6];
    const float* bih_b = (const float*)d_in[7];
    const float* bhh_b = (const float*)d_in[8];
    const float* Wg    = (const float*)d_in[9];
    const float* bg    = (const float*)d_in[10];
    float* out = (float*)d_out;

    char* ws = (char*)d_ws;
    float*    xg_f   = (float*)(ws + OFF_XG_F);
    float*    xg_b   = (float*)(ws + OFF_XG_B);
    float*    hf32   = (float*)(ws + OFF_HF32);
    float*    hb32   = (float*)(ws + OFF_HB32);
    unsigned short* hfbf = (unsigned short*)(ws + OFF_HFBF);
    unsigned short* hbbf = (unsigned short*)(ws + OFF_HBBF);
    unsigned* xpack  = (unsigned*)(ws + OFF_XBF);
    unsigned* wih_fp = (unsigned*)(ws + OFF_WIH_F);
    unsigned* wih_bp = (unsigned*)(ws + OFF_WIH_B);
    unsigned* whh_fp = (unsigned*)(ws + OFF_WHH_F);
    unsigned* whh_bp = (unsigned*)(ws + OFF_WHH_B);
    unsigned* wg_p   = (unsigned*)(ws + OFF_WG_P);
    float*    biasp  = (float*)(ws + OFF_BIAS);
    unsigned short* hcur = (unsigned short*)(ws + OFF_HCUR);
    float*    ccur   = (float*)(ws + OFF_CCUR);
    unsigned* bars   = (unsigned*)(ws + OFF_BARS);

    init_kernel<<<256, 256, 0, stream>>>((unsigned*)hcur, (unsigned*)ccur, bars);
    pack_x_kernel<<<32768, 256, 0, stream>>>(x, xpack);
    pack_w4_kernel<<<8192, 256, 0, stream>>>(Wih_f, wih_fp);
    pack_w4_kernel<<<8192, 256, 0, stream>>>(Wih_b, wih_bp);
    pack_w4_kernel<<<8192, 256, 0, stream>>>(Whh_f, whh_fp);
    pack_w4_kernel<<<8192, 256, 0, stream>>>(Whh_b, whh_bp);
    pack_wg_kernel<<<4096, 256, 0, stream>>>(Wg, wg_p);
    bias_kernel<<<32, 256, 0, stream>>>(bih_f, bhh_f, bih_b, bhh_b, biasp);

    input_gemm_kernel<<<dim3(1024, 8, 2), 256, 0, stream>>>(xpack, wih_fp, wih_bp, biasp,
                                                            xg_f, xg_b);
    scan_kernel<<<dim3(32, 2), 256, 0, stream>>>(xg_f, xg_b, whh_fp, whh_bp,
                                                 hcur, ccur, hf32, hb32, hfbf, hbbf, bars);
    gate_kernel<<<dim3(1024, 8), 256, 0, stream>>>(hfbf, hbbf, wg_p, bg, hf32, hb32, out);
}
